// Model_73813307949275
// MI455X (gfx1250) — matrix operand order repaired and checked by reading, not compiled, not run on hardware
//
#include <hip/hip_runtime.h>

// Flash-attention (fwd) with dropout for B=4,H=16,S=2048,D=64, f32 in/out.
// Matmuls run through v_wmma_f32_16x16x32_f16 (f32 accumulate).
// One workgroup = 4 waves = 64 query rows of one (b,h) head.
// K is staged row-major in LDS (B-frags = contiguous ds_load_b128);
// V is staged TRANSPOSED in LDS (d-major, padded stride) so its B-frags
// are also contiguous LDS reads. P is re-swizzled C->A layout via LDS.

typedef __attribute__((ext_vector_type(16))) _Float16 v16h;
typedef __attribute__((ext_vector_type(8)))  float    v8f;
typedef __attribute__((ext_vector_type(4)))  float    f4;
typedef __attribute__((ext_vector_type(4)))  _Float16 h4;

#define S_LEN 2048
#define DHEAD 64
#define QBLK  64    // query rows per workgroup (16 per wave)
#define KVBLK 64    // kv columns per main-loop iteration (4 x 16 tiles)
#define VSTRIDE 68  // padded row stride (halves) for transposed V tile

__device__ __forceinline__ unsigned pcg_hash(unsigned x) {
    // cheap 32-bit hash (PCG output permutation) for dropout mask bytes
    x = x * 747796405u + 2891336453u;
    x = ((x >> ((x >> 28) + 4u)) ^ x) * 277803737u;
    return x ^ (x >> 22);
}

__global__ __launch_bounds__(128) void fattn_fwd_kernel(
    const float* __restrict__ Q, const float* __restrict__ K,
    const float* __restrict__ V, float* __restrict__ O) {

    __shared__ _Float16 k_smem[KVBLK * DHEAD];       // K tile, row-major   8KB
    __shared__ _Float16 vt_smem[DHEAD * VSTRIDE];    // V tile, transposed ~8.5KB
    __shared__ _Float16 p_smem[4 * 16 * KVBLK];      // per-wave P tiles    8KB

    const int bh   = blockIdx.y;
    const int qb   = blockIdx.x * QBLK;
    const int tid  = threadIdx.x;
    const int wave = tid >> 5;
    const int lane = tid & 31;
    const int l16  = lane & 15;
    const bool hi  = (lane >= 16);

    const size_t head_off = (size_t)bh * S_LEN * DHEAD;
    const int qrow0 = qb + wave * 16;

    // ---- Q tile (16x64 f32) -> two 16x32 f16 A-fragments (resident) ----
    // A 16-bit 16x32 layout: M = lane%16; lanes<16 hold K={0..7,16..23},
    // lanes>=16 hold K={8..15,24..31} (element e -> K = e<8 ? e : e+8, +8 for hi).
    v16h aq[2];
    {
        const float* qrow = Q + head_off + (size_t)(qrow0 + l16) * DHEAD;
        #pragma unroll
        for (int c = 0; c < 2; ++c) {
            const int base = c * 32 + (hi ? 8 : 0);
            #pragma unroll
            for (int e = 0; e < 8; ++e) aq[c][e]     = (_Float16)qrow[base + e];
            #pragma unroll
            for (int e = 0; e < 8; ++e) aq[c][8 + e] = (_Float16)qrow[base + 16 + e];
        }
    }

    v8f oacc[4] = {};                 // O accumulator: 16x64 (4 N-tiles of 16)
    float mrow[8], lrow[8];           // online-softmax row state (UNSCALED max)
    #pragma unroll
    for (int r = 0; r < 8; ++r) { mrow[r] = -3.0e37f; lrow[r] = 0.0f; }

    _Float16* pm = p_smem + wave * (16 * KVBLK);

    for (int kv0 = 0; kv0 < S_LEN; kv0 += KVBLK) {
        __syncthreads();   // previous iteration's LDS reads complete

        // ---- cooperative K/V tile loads: 64x64 f32 -> f16 in LDS ----
        // K stored row-major; V stored transposed (vT[d][kv], stride 68).
        {
            const float* kbase = K + head_off + (size_t)kv0 * DHEAD;
            const float* vbase = V + head_off + (size_t)kv0 * DHEAD;
            #pragma unroll
            for (int j = 0; j < 8; ++j) {
                const int e4 = j * 128 + tid;            // float4 index (1024 total)
                const f4 ks = *(const f4*)(kbase + (size_t)e4 * 4);
                const f4 vs = *(const f4*)(vbase + (size_t)e4 * 4);
                *(h4*)(k_smem + e4 * 4) = __builtin_convertvector(ks, h4);
                const h4 vh = __builtin_convertvector(vs, h4);
                const int kvrow = e4 >> 4;               // kv row of this float4
                const int dbase = (e4 & 15) * 4;         // d of first element
                #pragma unroll
                for (int i = 0; i < 4; ++i)
                    vt_smem[(dbase + i) * VSTRIDE + kvrow] = vh[i];
            }
        }

        // prefetch next iteration's K/V tiles into cache
        if (kv0 + KVBLK < S_LEN) {
            __builtin_prefetch(V + head_off + (size_t)(kv0 + KVBLK) * DHEAD + tid * 32, 0, 1);
            __builtin_prefetch(K + head_off + (size_t)(kv0 + KVBLK) * DHEAD + tid * 32, 0, 1);
        }

        __syncthreads();   // K/V tiles visible

        // ---- S = Q @ K^T : four 16x16 score tiles, K-dim=64 (2 WMMAs each) ----
        // B 16-bit 32x16 layout: N = lane%16; lanes<16 K={0..7,16..23}, lanes>=16 K={8..15,24..31}.
        v8f sc[4];
        #pragma unroll
        for (int nt = 0; nt < 4; ++nt) {
            const _Float16* krow = k_smem + (nt * 16 + l16) * DHEAD;
            v8f c = {};
            #pragma unroll
            for (int kc = 0; kc < 2; ++kc) {
                v16h bk;
                const int dbase = kc * 32 + (hi ? 8 : 0);
                #pragma unroll
                for (int e = 0; e < 16; ++e) bk[e] = krow[dbase + (e < 8 ? e : e + 8)];
                c = __builtin_amdgcn_wmma_f32_16x16x32_f16(
                        false, aq[kc], false, bk, (short)0, c, false, false);
            }
            sc[nt] = c;   // raw qk; sqrt(D)=8 scale folded into softmax below
        }

        // ---- online softmax (unscaled-max domain): row max over 64 columns ----
        float rmax[8];
        #pragma unroll
        for (int r = 0; r < 8; ++r)
            rmax[r] = fmaxf(fmaxf(sc[0][r], sc[1][r]), fmaxf(sc[2][r], sc[3][r]));
        #pragma unroll
        for (int msk = 1; msk <= 8; msk <<= 1) {
            #pragma unroll
            for (int r = 0; r < 8; ++r)
                rmax[r] = fmaxf(rmax[r], __shfl_xor(rmax[r], msk, 32));
        }

        float alpha[8], negm8[8];
        #pragma unroll
        for (int r = 0; r < 8; ++r) {
            const float mn = fmaxf(mrow[r], rmax[r]);
            alpha[r] = __expf(8.0f * (mrow[r] - mn));   // rescale of old state
            mrow[r]  = mn;
            negm8[r] = -8.0f * mn;
        }

        // ---- p = exp(8*s - 8*m); accumulate pre-dropout row sums;
        //      write dropout-masked, 10x-rescaled p (f16) to per-wave LDS.
        //      One 32-bit hash yields 4 mask bytes (4 rows x this column). ----
        float rsum[8];
        #pragma unroll
        for (int r = 0; r < 8; ++r) rsum[r] = 0.0f;

        #pragma unroll
        for (int nt = 0; nt < 4; ++nt) {
            const int col = kv0 + nt * 16 + l16;
            #pragma unroll
            for (int g = 0; g < 2; ++g) {
                const unsigned rowbase = (unsigned)(qrow0 + (hi ? 8 : 0) + g * 4);
                const unsigned hh = pcg_hash((unsigned)bh * 4194304u +
                                             rowbase * 2048u + (unsigned)col + 42u);
                #pragma unroll
                for (int rr = 0; rr < 4; ++rr) {
                    const int r    = g * 4 + rr;
                    const int rowL = r + (hi ? 8 : 0);
                    const float pe = __expf(fmaf(8.0f, sc[nt][r], negm8[r]));
                    rsum[r] += pe;
                    const bool keep = ((hh >> (8 * rr)) & 255u) < 26u;  // ~= 0.1
                    pm[rowL * KVBLK + nt * 16 + l16] =
                        (_Float16)(keep ? pe * 10.0f : 0.0f);
                }
            }
        }
        #pragma unroll
        for (int msk = 1; msk <= 8; msk <<= 1) {
            #pragma unroll
            for (int r = 0; r < 8; ++r)
                rsum[r] += __shfl_xor(rsum[r], msk, 32);
        }
        #pragma unroll
        for (int r = 0; r < 8; ++r) lrow[r] = lrow[r] * alpha[r] + rsum[r];
        #pragma unroll
        for (int nt2 = 0; nt2 < 4; ++nt2) {
            #pragma unroll
            for (int r = 0; r < 8; ++r) oacc[nt2][r] *= alpha[r];
        }

        __syncthreads();   // P tiles visible (per-wave, but keep block in lockstep)

        // ---- O += P @ V : A = P (16x64, two 16x32 frags), B = V (32x16 tiles)
        //      from transposed LDS tile: two runs of 8 contiguous halves per lane. ----
        v16h pa[2];
        #pragma unroll
        for (int c = 0; c < 2; ++c) {
            #pragma unroll
            for (int e = 0; e < 16; ++e) {
                const int kk = c * 32 + (hi ? (e < 8 ? e + 8 : e + 16)
                                            : (e < 8 ? e     : e + 8));
                pa[c][e] = pm[l16 * KVBLK + kk];
            }
        }
        #pragma unroll
        for (int nt2 = 0; nt2 < 4; ++nt2) {
            const _Float16* vtrow = vt_smem + (nt2 * 16 + l16) * VSTRIDE;
            #pragma unroll
            for (int c = 0; c < 2; ++c) {
                v16h vb;
                const int kb = c * 32 + (hi ? 8 : 0);
                #pragma unroll
                for (int e = 0; e < 16; ++e) vb[e] = vtrow[kb + (e < 8 ? e : e + 8)];
                oacc[nt2] = __builtin_amdgcn_wmma_f32_16x16x32_f16(
                                false, pa[c], false, vb, (short)0, oacc[nt2], false, false);
            }
        }
    }

    // ---- normalize by softmax denominator and store O (f32) ----
    #pragma unroll
    for (int r = 0; r < 8; ++r) {
        const float inv = 1.0f / lrow[r];
        const int row   = qrow0 + r + (hi ? 8 : 0);
        float* orow = O + head_off + (size_t)row * DHEAD;
        #pragma unroll
        for (int nt2 = 0; nt2 < 4; ++nt2)
            orow[nt2 * 16 + l16] = oacc[nt2][r] * inv;
    }
}

extern "C" void kernel_launch(void* const* d_in, const int* in_sizes, int n_in,
                              void* d_out, int out_size, void* d_ws, size_t ws_size,
                              hipStream_t stream) {
    (void)in_sizes; (void)n_in; (void)out_size; (void)d_ws; (void)ws_size;
    const float* q = (const float*)d_in[0];
    const float* k = (const float*)d_in[1];
    const float* v = (const float*)d_in[2];
    float* o = (float*)d_out;
    dim3 grid(S_LEN / QBLK, 4 * 16);   // (query blocks, B*H)
    fattn_fwd_kernel<<<grid, 128, 0, stream>>>(q, k, v, o);
}
